// PointTransformerLayer_83219286327741
// MI455X (gfx1250) — matrix operand order repaired and checked by reading, not compiled, not run on hardware
//
#include <hip/hip_runtime.h>

// ---------------------------------------------------------------------------
// PointTransformerLayer for MI455X (gfx1250, wave32, WMMA + TDM)
// b=8, n=2048, k=16, emb=256, INNER=64  -> 16384 points
// GEMM work on v_wmma_f32_16x16x32_bf16; weights stored transposed (K-major)
// so every fragment is two contiguous b128 loads; pos staging via TDM.
// ---------------------------------------------------------------------------

typedef __attribute__((ext_vector_type(16))) __bf16 v16bf;
typedef __attribute__((ext_vector_type(8)))  float  v8f;
typedef __attribute__((ext_vector_type(4)))  unsigned int su4;
typedef __attribute__((ext_vector_type(8)))  unsigned int su8;

#define NPTS   2048
#define NBATCH 8
#define NPOINT (NPTS * NBATCH)   // 16384
#define KNN    16
#define EMB    256
#define INNER  64

__device__ __forceinline__ v8f wmma_bf16(v16bf a, v16bf b, v8f c) {
  return __builtin_amdgcn_wmma_f32_16x16x32_bf16(false, a, false, b,
                                                 (short)0, c, false, false);
}

// A-fragment (16x32 bf16, M x K), row-major A. Lane lm = M row, kh = K-half.
// element e -> K = kb + (e&7) + 16*(e>>3) + 8*kh  => two contiguous 8-runs.
__device__ __forceinline__ v16bf load_a(const __bf16* base, int ld, int kb,
                                        int lm, int kh) {
  const __bf16* p = base + (size_t)lm * ld + kb + 8 * kh;
  v16bf a;
#pragma unroll
  for (int e = 0; e < 16; ++e) a[e] = p[(e & 7) + 16 * (e >> 3)];
  return a;
}

// B-fragment (32x16 bf16, K x N) from TRANSPOSED weights Bt[N][K].
// lane lm = N column; element e -> K = kb + (e&7) + 16*(e>>3) + 8*kh  => two contiguous 8-runs.
__device__ __forceinline__ v16bf load_bt(const __bf16* Bt, int K, int kb,
                                         int n0, int lm, int kh) {
  const __bf16* p = Bt + (size_t)(n0 + lm) * K + kb + 8 * kh;
  v16bf b;
#pragma unroll
  for (int e = 0; e < 16; ++e) b[e] = p[(e & 7) + 16 * (e >> 3)];
  return b;
}

// ---------------------------------------------------------------------------
// Kernel 0a/0b: f32 -> bf16 packing (plain, and transposing K x N -> N x K)
// ---------------------------------------------------------------------------
__global__ void pack_bf16(const float* __restrict__ in, __bf16* __restrict__ out,
                          int n) {
  int i = blockIdx.x * blockDim.x + threadIdx.x;
  if (i < n) out[i] = (__bf16)in[i];
}

__global__ void pack_bf16_t(const float* __restrict__ in,
                            __bf16* __restrict__ out, int K, int N) {
  int i = blockIdx.x * blockDim.x + threadIdx.x;
  if (i < K * N) {
    const int k = i / N, n = i % N;
    out[(size_t)n * K + k] = (__bf16)in[i];
  }
}

// ---------------------------------------------------------------------------
// Kernel 1: kNN (top-k of sim = 2<pi,pj> - |pi|^2 - |pj|^2), one wave/query.
// Batch pos staged into LDS by the Tensor Data Mover (one descriptor).
// ---------------------------------------------------------------------------
__global__ __launch_bounds__(256) void knn_kernel(const float* __restrict__ pos,
                                                  int* __restrict__ idx) {
  __shared__ __align__(16) float ps[NPTS * 3];
  __shared__ float d2s[NPTS];

  const int qbase = blockIdx.x * 8;          // 8 waves per block
  const int batch = qbase / NPTS;            // 8 | 2048 -> no straddle
  const float* pb = pos + (size_t)batch * NPTS * 3;

  if (threadIdx.x < 32) {
    // TDM descriptor: 1-D tile of 6144 f32 (24 KB) global -> LDS.
    const unsigned lds = (unsigned)(uintptr_t)(&ps[0]);
    const unsigned long long ga = (unsigned long long)(uintptr_t)pb;
    su4 g0;
    g0[0] = 1u;                                    // count=1, user mode
    g0[1] = lds;                                   // lds_addr
    g0[2] = (unsigned)(ga & 0xffffffffu);          // global_addr[31:0]
    g0[3] = (unsigned)((ga >> 32) & 0x01ffffffu)   // global_addr[56:32]
            | (2u << 30);                          // type=2 ("image")
    su8 g1;
    g1[0] = (2u << 16);                            // data_size=4B, no flags
    g1[1] = (6144u & 0xffffu) << 16;               // tensor_dim0 lo16
    g1[2] = 0u;                                    // tensor_dim0 hi16, dim1 lo
    g1[3] = (6144u << 16);                         // tile_dim0 = 6144
    g1[4] = 0u;                                    // tile_dim1/2 unused
    g1[5] = 6144u;                                 // tensor_dim0_stride lo32
    g1[6] = 0u;
    g1[7] = 0u;
    asm volatile("tensor_load_to_lds %0, %1" : : "s"(g0), "s"(g1) : "memory");
    __builtin_amdgcn_s_wait_tensorcnt(0);
  }
  __syncthreads();

  for (int u = threadIdx.x; u < NPTS; u += blockDim.x)
    d2s[u] = ps[u * 3] * ps[u * 3] + ps[u * 3 + 1] * ps[u * 3 + 1] +
             ps[u * 3 + 2] * ps[u * 3 + 2];
  __syncthreads();

  const int w    = threadIdx.x >> 5;
  const int lane = threadIdx.x & 31;
  const int i    = (qbase % NPTS) + w;       // local query index
  const float qx = ps[i * 3], qy = ps[i * 3 + 1], qz = ps[i * 3 + 2];
  const float qd = d2s[i];

  float simv[64];
#pragma unroll
  for (int u = 0; u < 64; ++u) {
    const int j = u * 32 + lane;
    simv[u] = 2.0f * (qx * ps[j * 3] + qy * ps[j * 3 + 1] + qz * ps[j * 3 + 2]) -
              qd - d2s[j];
  }

  for (int r = 0; r < KNN; ++r) {
    float best  = -3.4e38f;
    int   bestj = NPTS;
#pragma unroll
    for (int u = 0; u < 64; ++u) {
      const int j = u * 32 + lane;
      const float v = simv[u];
      if (v > best || (v == best && j < bestj)) { best = v; bestj = j; }
    }
#pragma unroll
    for (int off = 16; off > 0; off >>= 1) {
      const float ov = __shfl_xor(best, off, 32);
      const int   oj = __shfl_xor(bestj, off, 32);
      if (ov > best || (ov == best && oj < bestj)) { best = ov; bestj = oj; }
    }
    if (lane == 0)
      idx[((size_t)batch * NPTS + i) * KNN + r] = batch * NPTS + bestj;
    if ((bestj & 31) == lane) {
#pragma unroll
      for (int u = 0; u < 64; ++u)
        if (u == (bestj >> 5)) simv[u] = -3.4e38f;
    }
  }
}

// ---------------------------------------------------------------------------
// Kernel 2/5: bf16 WMMA GEMM, one wave computes a 16x64 strip (4 n-tiles
// share each A fragment).  C[M,N](f32) = A[M,K] @ Bt[N,K]^T (+bias)
// ---------------------------------------------------------------------------
__global__ __launch_bounds__(128) void gemm_bf16_wave(
    const __bf16* __restrict__ A, const __bf16* __restrict__ Bt,
    const float* __restrict__ bias, float* __restrict__ C, int M, int N, int K) {
  const int wave = (blockIdx.x * blockDim.x + threadIdx.x) >> 5;
  const int lane = threadIdx.x & 31;
  const int ngrp = N >> 6;                   // groups of 64 output columns
  const int mt   = wave / ngrp;
  if (mt * 16 >= M) return;
  const int m0 = mt * 16, n0 = (wave % ngrp) * 64;
  const int lm = lane & 15, kh = lane >> 4;

  v8f acc[4] = {};
  for (int kb = 0; kb < K; kb += 32) {
    const __bf16* arow = A + (size_t)(m0 + lm) * K + kb + 8 * kh;
    __builtin_prefetch(arow + 64, 0, 1);
    v16bf a;
#pragma unroll
    for (int e = 0; e < 16; ++e) a[e] = arow[(e & 7) + 16 * (e >> 3)];
#pragma unroll
    for (int t = 0; t < 4; ++t)
      acc[t] = wmma_bf16(a, load_bt(Bt, K, kb, n0 + t * 16, lm, kh), acc[t]);
  }
#pragma unroll
  for (int t = 0; t < 4; ++t) {
    const int c = n0 + t * 16 + lm;
    const float bs = bias ? bias[c] : 0.0f;
#pragma unroll
    for (int r = 0; r < 8; ++r)
      C[(size_t)(m0 + r + 8 * kh) * N + c] = acc[t][r] + bs;
  }
}

// ---------------------------------------------------------------------------
// Kernel 4: fused per-point attention. One wave32 per point, 2 waves/block.
// ---------------------------------------------------------------------------
__global__ __launch_bounds__(64) void ptl_point_kernel(
    const float* __restrict__ pos, const float* __restrict__ qkv,
    const int* __restrict__ nbr_idx, const float* __restrict__ w_pos1,
    const float* __restrict__ b_pos1, const __bf16* __restrict__ wpos2T,
    const float* __restrict__ b_pos2, const __bf16* __restrict__ watt1T,
    const float* __restrict__ b_att1, const __bf16* __restrict__ watt2T,
    const float* __restrict__ b_att2, __bf16* __restrict__ agg_bf) {
  __shared__ int    nbr[2][KNN];
  __shared__ __align__(16) __bf16 h16[2][KNN][INNER];   // hidden, then h
  __shared__ float  f32a[2][KNN][INNER];                // rpe, then sim/attn
  __shared__ float  vpr[2][KNN][INNER];                 // v_nn + rpe
  __shared__ __align__(16) __bf16 act16[2][KNN][4 * INNER];

  const int w    = threadIdx.x >> 5;
  const int lane = threadIdx.x & 31;
  const int lm   = lane & 15, kh = lane >> 4;
  const int p    = blockIdx.x * 2 + w;

  if (lane < KNN) nbr[w][lane] = nbr_idx[(size_t)p * KNN + lane];
  __syncthreads();

  // --- pos MLP hidden: relu(rel_pos @ w_pos1 + b_pos1)
  {
    const int t = lm;
    const int j = nbr[w][t];
    const float qx = pos[(size_t)p * 3], qy = pos[(size_t)p * 3 + 1],
                qz = pos[(size_t)p * 3 + 2];
    const float rx = pos[(size_t)j * 3] - qx, ry = pos[(size_t)j * 3 + 1] - qy,
                rz = pos[(size_t)j * 3 + 2] - qz;
#pragma unroll
    for (int u = 0; u < 32; ++u) {
      const int c = kh * 32 + u;
      const float hv =
          b_pos1[c] + rx * w_pos1[c] + ry * w_pos1[64 + c] + rz * w_pos1[128 + c];
      h16[w][t][c] = (__bf16)fmaxf(hv, 0.0f);
    }
  }
  __syncthreads();

  // --- rpe = hidden(16x64) @ w_pos2(64x64) + b_pos2
  {
    const v16bf a0 = load_a(&h16[w][0][0], INNER, 0, lm, kh);
    const v16bf a1 = load_a(&h16[w][0][0], INNER, 32, lm, kh);
#pragma unroll
    for (int nt = 0; nt < 4; ++nt) {
      v8f acc = {};
      acc = wmma_bf16(a0, load_bt(wpos2T, INNER, 0, nt * 16, lm, kh), acc);
      acc = wmma_bf16(a1, load_bt(wpos2T, INNER, 32, nt * 16, lm, kh), acc);
      const int c = nt * 16 + lm;
      const float bs = b_pos2[c];
#pragma unroll
      for (int r = 0; r < 8; ++r) f32a[w][r + 8 * kh][c] = acc[r] + bs;
    }
  }
  __syncthreads();

  // --- gather q/k/v; h = q-k+rpe (bf16), vpr = v+rpe (f32)
#pragma unroll
  for (int u = 0; u < 32; ++u) {
    const int e = u * 32 + lane;
    const int t = e >> 6, c = e & 63;
    const int j = nbr[w][t];
    const float* base = qkv + (size_t)j * 192;
    const float rp = f32a[w][t][c];
    h16[w][t][c] = (__bf16)(base[c] - base[64 + c] + rp);
    vpr[w][t][c] = base[128 + c] + rp;
  }
  __syncthreads();

  // --- att layer1: relu(h(16x64) @ w_att1(64x256) + b_att1) -> act16
  {
    const v16bf a0 = load_a(&h16[w][0][0], INNER, 0, lm, kh);
    const v16bf a1 = load_a(&h16[w][0][0], INNER, 32, lm, kh);
#pragma unroll
    for (int nt = 0; nt < 16; ++nt) {
      v8f acc = {};
      acc = wmma_bf16(a0, load_bt(watt1T, INNER, 0, nt * 16, lm, kh), acc);
      acc = wmma_bf16(a1, load_bt(watt1T, INNER, 32, nt * 16, lm, kh), acc);
      const int c = nt * 16 + lm;
      const float bs = b_att1[c];
#pragma unroll
      for (int r = 0; r < 8; ++r)
        act16[w][r + 8 * kh][c] = (__bf16)fmaxf(acc[r] + bs, 0.0f);
    }
  }
  __syncthreads();

  // --- att layer2: sim = act(16x256) @ w_att2(256x64) + b_att2 -> f32a
  {
    v8f acc2[4] = {};
#pragma unroll
    for (int ks = 0; ks < 8; ++ks) {
      const v16bf a = load_a(&act16[w][0][0], 4 * INNER, ks * 32, lm, kh);
#pragma unroll
      for (int nt = 0; nt < 4; ++nt)
        acc2[nt] = wmma_bf16(
            a, load_bt(watt2T, 4 * INNER, ks * 32, nt * 16, lm, kh), acc2[nt]);
    }
#pragma unroll
    for (int nt = 0; nt < 4; ++nt) {
      const int c = nt * 16 + lm;
      const float bs = b_att2[c];
#pragma unroll
      for (int r = 0; r < 8; ++r) f32a[w][r + 8 * kh][c] = acc2[nt][r] + bs;
    }
  }
  __syncthreads();

  // --- softmax over channel dim (64) per neighbor; lane pair (lm, lm+16)
  {
    const int t = lm;
    float m = -3.4e38f;
#pragma unroll
    for (int u = 0; u < 32; ++u) m = fmaxf(m, f32a[w][t][kh * 32 + u]);
    m = fmaxf(m, __shfl_xor(m, 16, 32));
    float s = 0.0f;
#pragma unroll
    for (int u = 0; u < 32; ++u) s += __expf(f32a[w][t][kh * 32 + u] - m);
    s += __shfl_xor(s, 16, 32);
    const float inv = 1.0f / s;
#pragma unroll
    for (int u = 0; u < 32; ++u) {
      const int c = kh * 32 + u;
      f32a[w][t][c] = __expf(f32a[w][t][c] - m) * inv;
    }
  }
  __syncthreads();

  // --- L2 normalize over neighbor dim, aggregate
#pragma unroll
  for (int hh = 0; hh < 2; ++hh) {
    const int c = hh * 32 + lane;
    float ss = 0.0f;
#pragma unroll
    for (int t = 0; t < KNN; ++t) {
      const float a = f32a[w][t][c];
      ss += a * a;
    }
    const float inv = 1.0f / fmaxf(sqrtf(ss), 1e-12f);
    float ag = 0.0f;
#pragma unroll
    for (int t = 0; t < KNN; ++t) ag += f32a[w][t][c] * inv * vpr[w][t][c];
    agg_bf[(size_t)p * INNER + c] = (__bf16)ag;
  }
}

// ---------------------------------------------------------------------------
// Launch
// ---------------------------------------------------------------------------
extern "C" void kernel_launch(void* const* d_in, const int* in_sizes, int n_in,
                              void* d_out, int out_size, void* d_ws,
                              size_t ws_size, hipStream_t stream) {
  (void)in_sizes; (void)n_in; (void)out_size; (void)ws_size;

  const float* x      = (const float*)d_in[0];
  const float* pos    = (const float*)d_in[1];
  const float* w_qkv  = (const float*)d_in[2];
  const float* w_pos1 = (const float*)d_in[3];
  const float* b_pos1 = (const float*)d_in[4];
  const float* w_pos2 = (const float*)d_in[5];
  const float* b_pos2 = (const float*)d_in[6];
  const float* w_att1 = (const float*)d_in[7];
  const float* b_att1 = (const float*)d_in[8];
  const float* w_att2 = (const float*)d_in[9];
  const float* b_att2 = (const float*)d_in[10];
  const float* w_out  = (const float*)d_in[11];
  const float* b_out  = (const float*)d_in[12];

  char* ws = (char*)d_ws;
  size_t off = 0;
  __bf16* x_bf   = (__bf16*)(ws + off); off += (size_t)NPOINT * EMB * 2;   // 8 MB
  __bf16* wqkvT  = (__bf16*)(ws + off); off += (size_t)EMB * 192 * 2;      // [192][256]
  __bf16* wpos2T = (__bf16*)(ws + off); off += (size_t)INNER * INNER * 2;  // [64][64]
  __bf16* watt1T = (__bf16*)(ws + off); off += (size_t)INNER * 4 * INNER * 2; // [256][64]
  __bf16* watt2T = (__bf16*)(ws + off); off += (size_t)4 * INNER * INNER * 2; // [64][256]
  __bf16* woutT  = (__bf16*)(ws + off); off += (size_t)INNER * EMB * 2;    // [256][64]
  float*  qkv    = (float*)(ws + off);  off += (size_t)NPOINT * 192 * 4;   // 12 MB
  int*    idxb   = (int*)(ws + off);    off += (size_t)NPOINT * KNN * 4;   // 1 MB
  __bf16* agg_bf = (__bf16*)(ws + off); off += (size_t)NPOINT * INNER * 2; // 2 MB

  // 0) pack to bf16 (weights transposed to K-major for fragment loads)
  {
    int n;
    n = NPOINT * EMB;
    pack_bf16<<<(n + 255) / 256, 256, 0, stream>>>(x, x_bf, n);
    n = EMB * 192;
    pack_bf16_t<<<(n + 255) / 256, 256, 0, stream>>>(w_qkv, wqkvT, EMB, 192);
    n = INNER * INNER;
    pack_bf16_t<<<(n + 255) / 256, 256, 0, stream>>>(w_pos2, wpos2T, INNER, INNER);
    n = INNER * 4 * INNER;
    pack_bf16_t<<<(n + 255) / 256, 256, 0, stream>>>(w_att1, watt1T, INNER, 4 * INNER);
    n = 4 * INNER * INNER;
    pack_bf16_t<<<(n + 255) / 256, 256, 0, stream>>>(w_att2, watt2T, 4 * INNER, INNER);
    n = INNER * EMB;
    pack_bf16_t<<<(n + 255) / 256, 256, 0, stream>>>(w_out, woutT, INNER, EMB);
  }

  // 1) kNN indices (flat row offsets, includes self)
  knn_kernel<<<NPOINT / 8, 256, 0, stream>>>(pos, idxb);

  // 2) qkv = x @ w_qkv : (16384x256)@(256x192)
  {
    const int waves = (NPOINT / 16) * (192 / 64);
    gemm_bf16_wave<<<(waves * 32 + 127) / 128, 128, 0, stream>>>(
        x_bf, wqkvT, nullptr, qkv, NPOINT, 192, EMB);
  }

  // 3) fused per-point attention -> agg (bf16)
  ptl_point_kernel<<<NPOINT / 2, 64, 0, stream>>>(
      pos, qkv, idxb, w_pos1, b_pos1, wpos2T, b_pos2, watt1T, b_att1, watt2T,
      b_att2, agg_bf);

  // 4) out = agg @ w_out + b_out : (16384x64)@(64x256)
  {
    const int waves = (NPOINT / 16) * (EMB / 64);
    gemm_bf16_wave<<<(waves * 32 + 127) / 128, 128, 0, stream>>>(
        agg_bf, woutT, b_out, (float*)d_out, NPOINT, EMB, INNER);
  }
}
